// WindowAttention_10496900071754
// MI455X (gfx1250) — matrix operand order repaired and checked by reading, not compiled, not run on hardware
//
#include <hip/hip_runtime.h>

typedef unsigned short u16;
typedef unsigned int   u32;
typedef __attribute__((ext_vector_type(16))) __bf16 v16bf;
typedef __attribute__((ext_vector_type(8)))  float  v8f;

#define NTOK   49
#define SCALE  0.17677669529663687f  /* 32^-0.5 */

struct U8x4 { u32 u[8]; };

__device__ __forceinline__ u16 f2bf(float f) {
  u32 u = __float_as_uint(f);
  u32 r = (u + 0x7FFFu + ((u >> 16) & 1u)) >> 16;
  return (u16)r;
}

__device__ __forceinline__ v8f zero8() {
  v8f z;
#pragma unroll
  for (int i = 0; i < 8; ++i) z[i] = 0.0f;
  return z;
}

// A-fragment, bf16 16x32 (MxK): lane m = lane&15; lanes0-15 K-halves {0-7,16-23},
// lanes16-31 {8-15,24-31}; VGPR j holds one K-pair (consecutive K -> one b32 load).
__device__ __forceinline__ v16bf ld_afrag(const u16* base, int row0, int pitch,
                                          int k0, int lane) {
  int m  = row0 + (lane & 15);
  int kh = (lane >> 4) << 3;  // 0 or 8
  const u16* r = base + m * pitch + k0;
  U8x4 t;
#pragma unroll
  for (int j = 0; j < 8; ++j) {
    int k = ((j & 4) << 2) + kh + ((j & 3) << 1);  // j<4 -> +0, j>=4 -> +16
    t.u[j] = *(const u32*)(r + k);
  }
  return __builtin_bit_cast(v16bf, t);
}

// B-fragment, bf16 32x16 (KxN): lane n = lane&15; lanes0-15 K {0-7,16-23}, lanes16-31
// K {8-15,24-31} (matches A-fragment); VGPR j holds one K-pair. "base" rows indexed by N, row holds
// consecutive K (i.e. the K^T / transposed-weight layout) -> contiguous b32 loads.
__device__ __forceinline__ v16bf ld_bfrag(const u16* base, int nrow0, int pitch,
                                          int k0, int lane) {
  int n  = nrow0 + (lane & 15);
  int kh = (lane >> 4) << 3;  // 0 or 8
  const u16* r = base + n * pitch + k0 + kh;
  U8x4 t;
#pragma unroll
  for (int j = 0; j < 8; ++j) t.u[j] = *(const u32*)(r + ((j & 4) << 2) + ((j & 3) << 1));
  return __builtin_bit_cast(v16bf, t);
}

__device__ __forceinline__ v8f wmma_bf(v16bf a, v16bf b, v8f c) {
  return __builtin_amdgcn_wmma_f32_16x16x32_bf16(false, a, false, b, (short)0, c,
                                                 false, false);
}

// ---- prep: transpose+convert weights to bf16, gather relative-position bias ----
__global__ void prep_kernel(const float* __restrict__ qkv_w,
                            const float* __restrict__ proj_w,
                            const float* __restrict__ bias_table,
                            const int*   __restrict__ rel_index,
                            u16* __restrict__ qkv_wt, u16* __restrict__ proj_wt,
                            float* __restrict__ relb) {
  int i = blockIdx.x * blockDim.x + threadIdx.x;
  int stride = gridDim.x * blockDim.x;
  for (int idx = i; idx < 768 * 256; idx += stride) {          // qkv_wt[o][c]
    int o = idx >> 8, c = idx & 255;
    qkv_wt[idx] = f2bf(qkv_w[c * 768 + o]);
  }
  for (int idx = i; idx < 256 * 256; idx += stride) {          // proj_wt[o][c]
    int o = idx >> 8, c = idx & 255;
    proj_wt[idx] = f2bf(proj_w[c * 256 + o]);
  }
  for (int idx = i; idx < 8 * 2401; idx += stride) {           // relb[h][i][j]
    int h = idx / 2401, r = idx % 2401;
    relb[idx] = bias_table[rel_index[r] * 8 + h];
  }
}

// ---- fused window attention: one block per window, one wave per head ----
__global__ __launch_bounds__(256, 1)
void swin_attn_kernel(const float* __restrict__ x, const float* __restrict__ mask,
                      const float* __restrict__ qkv_b,
                      const float* __restrict__ proj_b,
                      const u16* __restrict__ qkv_wt,
                      const u16* __restrict__ proj_wt,
                      const float* __restrict__ relb, float* __restrict__ out) {
  // LDS map (u16 units): [0,33792): xs[64][264] (phase0/1) then P[8][64][66] (ph2/3)
  //                      [33792,51200): Q[8][64][34]
  //                      [51200,68608): K[8][64][34] (ph1/2) then O[64][264] (ph3/4)
  //                      [68608,85504): Vt[8][32][66]
  __shared__ __align__(16) u16 smem[85504];
  u16* xs  = smem;
  u16* Qs  = smem + 33792;
  u16* Ks  = smem + 51200;
  u16* Vts = smem + 68608;

  const int tid  = threadIdx.x;
  const int w    = tid >> 5;          // wave = head
  const int lane = tid & 31;
  const int ln   = lane & 15;
  const int kh8  = (lane >> 4) << 3;  // C-tile M offset for lanes 16-31
  const int blk  = blockIdx.x;

  // ---------------- phase 0: x -> bf16 LDS, zero-pad rows 49..63 --------------
  {
    const float4* xg = (const float4*)(x + (size_t)blk * 12544);
    for (int i = tid; i < 3136; i += 256) {
      float4 f = xg[i];
      int e = i << 2, row = e >> 8, col = e & 255;
      uint2 pk;
      pk.x = (u32)f2bf(f.x) | ((u32)f2bf(f.y) << 16);
      pk.y = (u32)f2bf(f.z) | ((u32)f2bf(f.w) << 16);
      *(uint2*)(&xs[row * 264 + col]) = pk;
    }
    for (int i = tid; i < 15 * 132; i += 256) {
      int r = 49 + i / 132, c = (i % 132) << 1;
      *(u32*)(&xs[r * 264 + c]) = 0u;
    }
    __syncthreads();
  }

  // -------- phase 1: qkv = x @ qkv_w + b; route to Q / K / V^T in LDS --------
  {
    const int cbase = w * 96;  // this wave's 96 output columns of 768
#pragma unroll
    for (int half = 0; half < 2; ++half) {
      v8f acc[12];
#pragma unroll
      for (int t = 0; t < 12; ++t) acc[t] = zero8();
#pragma unroll
      for (int kk = 0; kk < 8; ++kk) {
        v16bf a[4];
#pragma unroll
        for (int tm = 0; tm < 4; ++tm)
          a[tm] = ld_afrag(xs, tm * 16, 264, kk * 32, lane);
#pragma unroll
        for (int nt = 0; nt < 3; ++nt) {
          int col0 = cbase + (half * 3 + nt) * 16;
          v16bf b = ld_bfrag(qkv_wt, col0, 256, kk * 32, lane);
#pragma unroll
          for (int tm = 0; tm < 4; ++tm)
            acc[nt * 4 + tm] = wmma_bf(a[tm], b, acc[nt * 4 + tm]);
        }
      }
#pragma unroll
      for (int nt = 0; nt < 3; ++nt) {
        int col = cbase + (half * 3 + nt) * 16 + ln;
        float bias = qkv_b[col];
#pragma unroll
        for (int tm = 0; tm < 4; ++tm) {
#pragma unroll
          for (int g = 0; g < 8; ++g) {
            int m = tm * 16 + g + kh8;
            u16 hv = f2bf(acc[nt * 4 + tm][g] + bias);
            if (col < 256) {                       // Q[h][tok][d]
              int hh = col >> 5, d = col & 31;
              Qs[hh * 2176 + m * 34 + d] = hv;
            } else if (col < 512) {                // K[h][tok][d]
              int c2 = col - 256, hh = c2 >> 5, d = c2 & 31;
              Ks[hh * 2176 + m * 34 + d] = hv;
            } else {                               // Vt[h][d][tok]
              int c2 = col - 512, hh = c2 >> 5, d = c2 & 31;
              Vts[hh * 2112 + d * 66 + m] = hv;
            }
          }
        }
      }
    }
    __syncthreads();
  }

  const int h = w;
  u16* Pp = smem + w * 4224;  // per-wave P[64][66] bf16, overlays xs

  // ---- phase 2: S = Q Kt, +bias+mask, in-register softmax, P -> LDS bf16 ----
  {
    const u16* Qh = Qs + h * 2176;
    const u16* Kh = Ks + h * 2176;
    v16bf bq[4];
#pragma unroll
    for (int tn = 0; tn < 4; ++tn) bq[tn] = ld_bfrag(Kh, tn * 16, 34, 0, lane);
    v8f s[16];
#pragma unroll
    for (int tm = 0; tm < 4; ++tm) {
      v16bf a = ld_afrag(Qh, tm * 16, 34, 0, lane);
#pragma unroll
      for (int tn = 0; tn < 4; ++tn) s[tm * 4 + tn] = wmma_bf(a, bq[tn], zero8());
    }
    const int wi = blk & 63;
    const float* mw = mask + wi * 2401;
    const float* rb = relb + h * 2401;
#pragma unroll
    for (int tm = 0; tm < 4; ++tm)
#pragma unroll
      for (int tn = 0; tn < 4; ++tn)
#pragma unroll
        for (int g = 0; g < 8; ++g) {
          int m = tm * 16 + g + kh8;
          int n = tn * 16 + ln;
          float v = s[tm * 4 + tn][g];
          if (n < NTOK) {
            v *= SCALE;
            if (m < NTOK) v += rb[m * 49 + n] + mw[m * 49 + n];
          } else {
            v = -1e30f;  // pad columns -> exp() == 0
          }
          s[tm * 4 + tn][g] = v;
        }
    // softmax: row M of a C-tile lives across 16 lanes in one VGPR slot
#pragma unroll
    for (int tm = 0; tm < 4; ++tm) {
#pragma unroll
      for (int g = 0; g < 8; ++g) {
        float v0 = s[tm * 4 + 0][g], v1 = s[tm * 4 + 1][g];
        float v2 = s[tm * 4 + 2][g], v3 = s[tm * 4 + 3][g];
        float mx = fmaxf(fmaxf(v0, v1), fmaxf(v2, v3));
#pragma unroll
        for (int off = 8; off > 0; off >>= 1)
          mx = fmaxf(mx, __shfl_xor(mx, off, 16));
        float e0 = __expf(v0 - mx), e1 = __expf(v1 - mx);
        float e2 = __expf(v2 - mx), e3 = __expf(v3 - mx);
        float sum = e0 + e1 + e2 + e3;
#pragma unroll
        for (int off = 8; off > 0; off >>= 1) sum += __shfl_xor(sum, off, 16);
        float r = 1.0f / sum;
        int m = tm * 16 + g + kh8;
        Pp[m * 66 + ln]      = f2bf(e0 * r);
        Pp[m * 66 + 16 + ln] = f2bf(e1 * r);
        Pp[m * 66 + 32 + ln] = f2bf(e2 * r);
        Pp[m * 66 + 48 + ln] = f2bf(e3 * r);
      }
    }
    __syncthreads();  // all waves done reading K before O overlays it
  }

  // ------------------- phase 3: O_h = P @ V_h -> LDS bf16 --------------------
  {
    const u16* Vh = Vts + h * 2112;
    u16* O = Ks;  // overlay dead K region: O[tok][264]
    v16bf bv[2][2];
#pragma unroll
    for (int tn = 0; tn < 2; ++tn)
#pragma unroll
      for (int ks = 0; ks < 2; ++ks)
        bv[tn][ks] = ld_bfrag(Vh, tn * 16, 66, ks * 32, lane);
#pragma unroll
    for (int tm = 0; tm < 4; ++tm) {
      v16bf a0 = ld_afrag(Pp, tm * 16, 66, 0, lane);
      v16bf a1 = ld_afrag(Pp, tm * 16, 66, 32, lane);
#pragma unroll
      for (int tn = 0; tn < 2; ++tn) {
        v8f o = wmma_bf(a0, bv[tn][0], zero8());
        o = wmma_bf(a1, bv[tn][1], o);
#pragma unroll
        for (int g = 0; g < 8; ++g) {
          int m = tm * 16 + g + kh8;
          O[m * 264 + h * 32 + tn * 16 + ln] = f2bf(o[g]);
        }
      }
    }
    __syncthreads();
  }

  // --------------- phase 4: out = O @ proj_w + b  (f32 to global) ------------
  {
    const u16* O = Ks;
    const int cb = w * 32;  // this wave's 32 output columns of 256
    v8f acc[8];
#pragma unroll
    for (int t = 0; t < 8; ++t) acc[t] = zero8();
#pragma unroll
    for (int kk = 0; kk < 8; ++kk) {
      v16bf a[4];
#pragma unroll
      for (int tm = 0; tm < 4; ++tm)
        a[tm] = ld_afrag(O, tm * 16, 264, kk * 32, lane);
#pragma unroll
      for (int tn = 0; tn < 2; ++tn) {
        v16bf b = ld_bfrag(proj_wt, cb + tn * 16, 256, kk * 32, lane);
#pragma unroll
        for (int tm = 0; tm < 4; ++tm)
          acc[tm * 2 + tn] = wmma_bf(a[tm], b, acc[tm * 2 + tn]);
      }
    }
#pragma unroll
    for (int tn = 0; tn < 2; ++tn) {
      int col = cb + tn * 16 + ln;
      float pb = proj_b[col];
#pragma unroll
      for (int tm = 0; tm < 4; ++tm)
#pragma unroll
        for (int g = 0; g < 8; ++g) {
          int m = tm * 16 + g + kh8;
          if (m < NTOK)
            out[((size_t)blk * NTOK + m) * 256 + col] = acc[tm * 2 + tn][g] + pb;
        }
    }
  }
}

extern "C" void kernel_launch(void* const* d_in, const int* in_sizes, int n_in,
                              void* d_out, int out_size, void* d_ws, size_t ws_size,
                              hipStream_t stream) {
  (void)in_sizes; (void)n_in; (void)out_size; (void)ws_size;
  const float* x          = (const float*)d_in[0];
  const float* mask       = (const float*)d_in[1];
  const float* qkv_w      = (const float*)d_in[2];
  const float* qkv_b      = (const float*)d_in[3];
  const float* proj_w     = (const float*)d_in[4];
  const float* proj_b     = (const float*)d_in[5];
  const float* bias_table = (const float*)d_in[6];
  const int*   rel_index  = (const int*)d_in[7];

  // workspace: bf16 qkv_w^T [768][256] | bf16 proj_w^T [256][256] | relb f32 [8][49][49]
  u16*   qkv_wt  = (u16*)d_ws;
  u16*   proj_wt = (u16*)((char*)d_ws + 393216);
  float* relb    = (float*)((char*)d_ws + 524288);

  prep_kernel<<<512, 256, 0, stream>>>(qkv_w, proj_w, bias_table, rel_index,
                                       qkv_wt, proj_wt, relb);
  swin_attn_kernel<<<4096, 256, 0, stream>>>(x, mask, qkv_b, proj_b, qkv_wt,
                                             proj_wt, relb, (float*)d_out);
}
